// Hercules_25443386262247
// MI455X (gfx1250) — matrix operand order repaired and checked by reading, not compiled, not run on hardware
//
#include <hip/hip_runtime.h>
#include <hip/hip_bf16.h>

// ---------------------------------------------------------------------------
// Problem constants (match reference)
// ---------------------------------------------------------------------------
#define BATCH 4
#define SEQ   4096
#define DIM   2048
#define N3    (3 * DIM)          // 6144
#define MROWS (BATCH * SEQ)      // 16384
#define ROT_DIM (DIM / 2)        // 1024

// GEMM tiling
#define BM 128
#define BN 128
#define BK 32
#define ASTR (BK + 8)            // padded LDS row stride in ushorts (80 B)

typedef __attribute__((ext_vector_type(16))) __bf16 v16bf;
typedef __attribute__((ext_vector_type(8)))  float  v8f;

union AFrag { v16bf v; uint4 q[2]; };

// ---------------------------------------------------------------------------
// bf16 helpers (bit-level, round-to-nearest-even)
// ---------------------------------------------------------------------------
__device__ __forceinline__ unsigned short f2bf(float f) {
    unsigned int u = __float_as_uint(f);
    unsigned int r = u + 0x7FFFu + ((u >> 16) & 1u);
    return (unsigned short)(r >> 16);
}
__device__ __forceinline__ float bf2f(unsigned short h) {
    return __uint_as_float(((unsigned int)h) << 16);
}

// ---------------------------------------------------------------------------
// CDNA5 async global->LDS copy (ASYNCcnt-tracked, bypasses VGPRs).
// One call = 2 instructions = 32 bytes per lane. INST_OFFSET advances both
// the LDS and the global address (ISA 15.18.3), so one address pair suffices.
// ---------------------------------------------------------------------------
__device__ __forceinline__ void async_copy32(unsigned lds_off,
                                             const unsigned short* gaddr) {
    asm volatile("global_load_async_to_lds_b128 %0, %1, off\n\t"
                 "global_load_async_to_lds_b128 %0, %1, off offset:16"
                 :: "v"(lds_off), "v"(gaddr) : "memory");
}
__device__ __forceinline__ void wait_async_le4() {
    asm volatile("s_wait_asynccnt 0x4" ::: "memory");
}
__device__ __forceinline__ void wait_async_0() {
    asm volatile("s_wait_asynccnt 0x0" ::: "memory");
}

// ---------------------------------------------------------------------------
// Kernel 1: transpose + convert weights: src (K x N, f32) -> dst (N x K, bf16)
// ---------------------------------------------------------------------------
__global__ void __launch_bounds__(256)
k_wtrans_bf16(const float* __restrict__ src, unsigned short* __restrict__ dst,
              int K, int N) {
    size_t i = (size_t)blockIdx.x * 256 + threadIdx.x;
    if (i >= (size_t)K * N) return;
    int n = (int)(i / K);
    int k = (int)(i % K);
    dst[i] = f2bf(src[(size_t)k * N + n]);
}

// ---------------------------------------------------------------------------
// Kernel 2: LayerNorm, one block (256 thr = 8 waves) per row -> bf16 out
// ---------------------------------------------------------------------------
__global__ void __launch_bounds__(256)
k_layernorm(const float* __restrict__ X, const float* __restrict__ gamma,
            const float* __restrict__ beta, unsigned short* __restrict__ Xn) {
    const int row = blockIdx.x;
    const float* x = X + (size_t)row * DIM;

    float v[8];
    float s = 0.f, ss = 0.f;
#pragma unroll
    for (int i = 0; i < 8; ++i) {
        int d = threadIdx.x + i * 256;
        v[i] = x[d];
        s  += v[i];
        ss += v[i] * v[i];
    }
    // wave32 reduce
#pragma unroll
    for (int o = 16; o > 0; o >>= 1) {
        s  += __shfl_xor(s,  o, 32);
        ss += __shfl_xor(ss, o, 32);
    }
    __shared__ float smA[8], smB[8], bc[2];
    int lane = threadIdx.x & 31, w = threadIdx.x >> 5;
    if (lane == 0) { smA[w] = s; smB[w] = ss; }
    __syncthreads();
    if (threadIdx.x == 0) {
        float a = 0.f, b = 0.f;
#pragma unroll
        for (int i = 0; i < 8; ++i) { a += smA[i]; b += smB[i]; }
        bc[0] = a; bc[1] = b;
    }
    __syncthreads();
    float mu  = bc[0] * (1.0f / DIM);
    float var = bc[1] * (1.0f / DIM) - mu * mu;
    float rstd = rsqrtf(var + 1e-5f);

    unsigned short* out = Xn + (size_t)row * DIM;
#pragma unroll
    for (int i = 0; i < 8; ++i) {
        int d = threadIdx.x + i * 256;
        out[d] = f2bf((v[i] - mu) * rstd * gamma[d] + beta[d]);
    }
}

// ---------------------------------------------------------------------------
// Kernel 3/7: bf16 WMMA GEMM, C = A(MxK) * B^T(NxK) (+ epilogue)
//   mode 0: QKV epilogue (+b_in, split Q->bf16 / K,V->f32)
//   mode 1: out epilogue (+b_out + residual X -> f32 Y)
// Block: 256 thr = 8 waves, tile 128x128, k-step 32, double-buffered LDS
// filled with async global->LDS copies (ASYNCcnt pipeline).
// Wave w: rows [ (w&3)*32, +32 ), cols [ (w>>2)*64, +64 ) -> 2x4 16x16 accs.
// ---------------------------------------------------------------------------
__global__ void __launch_bounds__(256)
k_gemm_wmma(const unsigned short* __restrict__ Ag,   // M x K bf16 row-major
            const unsigned short* __restrict__ Bg,   // N x K bf16 row-major (B^T)
            const float* __restrict__ bias,          // N
            int Mdim, int Ndim, int Kdim, int mode,
            unsigned short* __restrict__ Qout,
            float* __restrict__ Kout,
            float* __restrict__ Vout,
            const float* __restrict__ Xres,
            float* __restrict__ Yout) {
    __shared__ unsigned short As[2][BM * ASTR];
    __shared__ unsigned short Bs[2][BN * ASTR];

    const int m0 = blockIdx.y * BM;
    const int n0 = blockIdx.x * BN;

    const int t    = threadIdx.x;
    const int lane = t & 31;
    const int wave = t >> 5;
    const int wm   = wave & 3;   // 0..3 : 32-row strip
    const int wn   = wave >> 2;  // 0..1 : 64-col strip
    const int l15  = lane & 15;
    const int lh   = lane >> 4;  // lane half

    const int fr = t >> 1;            // fill row 0..127
    const int fk = (t & 1) * 16;      // fill k-offset 0/16

    // per-thread fill addresses
    const unsigned short* gA = Ag + (size_t)(m0 + fr) * Kdim + fk;
    const unsigned short* gB = Bg + (size_t)(n0 + fr) * Kdim + fk;
    const unsigned ldsA[2] = {
        (unsigned)(unsigned long long)&As[0][fr * ASTR + fk],
        (unsigned)(unsigned long long)&As[1][fr * ASTR + fk] };
    const unsigned ldsB[2] = {
        (unsigned)(unsigned long long)&Bs[0][fr * ASTR + fk],
        (unsigned)(unsigned long long)&Bs[1][fr * ASTR + fk] };

    v8f acc[2][4] = {};

    const int nsteps = Kdim / BK;

    // prologue: stream tile 0 into buffer 0
    async_copy32(ldsA[0], gA);
    async_copy32(ldsB[0], gB);

    for (int i = 0; i < nsteps; ++i) {
        const int cur = i & 1;
        if (i + 1 < nsteps) {
            // stream tile i+1 into the other buffer (its last consumer was
            // iteration i-1, whose trailing barrier has been passed)
            async_copy32(ldsA[1 - cur], gA + (size_t)(i + 1) * BK);
            async_copy32(ldsB[1 - cur], gB + (size_t)(i + 1) * BK);
            wait_async_le4();          // tile i resident; tile i+1 in flight
        } else {
            wait_async_0();
        }
        __syncthreads();

        // ---- A fragments: 16x32 bf16 per ISA layout -----------------------
        // lane half lh: elements 0..7  <- K = lh*8 + 0..7
        //               elements 8..15 <- K = 16 + lh*8 + 0..7
        AFrag af[2];
#pragma unroll
        for (int mt = 0; mt < 2; ++mt) {
            int r = wm * 32 + mt * 16 + l15;
            af[mt].q[0] = *(const uint4*)&As[cur][r * ASTR + lh * 8];
            af[mt].q[1] = *(const uint4*)&As[cur][r * ASTR + 16 + lh * 8];
        }
        // ---- B fragments: 32x16 bf16; e<8 <- K = lh*8 + e, e>=8 <- K = 16 + lh*8 + e-8, N = l15 -
        AFrag bf_[4];
#pragma unroll
        for (int nt = 0; nt < 4; ++nt) {
            int c = wn * 64 + nt * 16 + l15;
            bf_[nt].q[0] = *(const uint4*)&Bs[cur][c * ASTR + lh * 8];
            bf_[nt].q[1] = *(const uint4*)&Bs[cur][c * ASTR + 16 + lh * 8];
        }
        // ---- 8 independent WMMAs per k-step ------------------------------
#pragma unroll
        for (int mt = 0; mt < 2; ++mt)
#pragma unroll
            for (int nt = 0; nt < 4; ++nt)
                acc[mt][nt] = __builtin_amdgcn_wmma_f32_16x16x32_bf16(
                    false, af[mt].v, false, bf_[nt].v,
                    (short)0, acc[mt][nt], false, false);
        __syncthreads();
    }

    // ---- epilogue: C/D layout: elem r -> M = r + 8*lh, N = l15 ------------
#pragma unroll
    for (int mt = 0; mt < 2; ++mt) {
#pragma unroll
        for (int nt = 0; nt < 4; ++nt) {
            int gcol = n0 + wn * 64 + nt * 16 + l15;
            float bval = bias[gcol];
#pragma unroll
            for (int r = 0; r < 8; ++r) {
                int grow = m0 + wm * 32 + mt * 16 + r + lh * 8;
                float val = acc[mt][nt][r] + bval;
                if (mode == 0) {
                    if (gcol < DIM)
                        Qout[(size_t)grow * DIM + gcol] = f2bf(val);
                    else if (gcol < 2 * DIM)
                        Kout[(size_t)grow * DIM + (gcol - DIM)] = val;
                    else
                        Vout[(size_t)grow * DIM + (gcol - 2 * DIM)] = val;
                } else {
                    size_t o = (size_t)grow * DIM + gcol;
                    Yout[o] = val + Xres[o];
                }
            }
        }
    }
}

// ---------------------------------------------------------------------------
// Kernel 4: rotary (in place, f32) + row L2 inverse-norm.
// grid (MROWS, 2): y==0 -> K buffer, y==1 -> V buffer. Block = 256.
// ---------------------------------------------------------------------------
__global__ void __launch_bounds__(256)
k_rotary_norm(float* __restrict__ Kf, float* __restrict__ Vf,
              float* __restrict__ rnorm /* [2*MROWS] */) {
    const int row   = blockIdx.x;
    const int which = blockIdx.y;
    float* base = (which == 0) ? Kf : Vf;
    float2* rowp = (float2*)(base + (size_t)row * DIM);
    const float s = (float)(row & (SEQ - 1));   // seq position

    float ss = 0.f;
#pragma unroll
    for (int i = 0; i < 4; ++i) {
        int j = threadIdx.x + i * 256;          // pair index, 0..1023
        float2 p = rowp[j];
        if (j < ROT_DIM / 2) {                  // rotary region: first 512 pairs
            float ang = s * __powf(10000.0f, -((float)(2 * j)) * (1.0f / ROT_DIM));
            float sn, cs;
            __sincosf(ang, &sn, &cs);
            float a = p.x * cs - p.y * sn;
            float b = p.y * cs + p.x * sn;
            p.x = a; p.y = b;
            rowp[j] = p;
        }
        ss += p.x * p.x + p.y * p.y;
    }
#pragma unroll
    for (int o = 16; o > 0; o >>= 1) ss += __shfl_xor(ss, o, 32);
    __shared__ float sm[8];
    int lane = threadIdx.x & 31, w = threadIdx.x >> 5;
    if (lane == 0) sm[w] = ss;
    __syncthreads();
    if (threadIdx.x == 0) {
        float a = 0.f;
#pragma unroll
        for (int i = 0; i < 8; ++i) a += sm[i];
        rnorm[which * MROWS + row] = rsqrtf(a);
    }
}

// ---------------------------------------------------------------------------
// Kernel 5a: zero A accumulator
// ---------------------------------------------------------------------------
__global__ void __launch_bounds__(256)
k_zero(float* __restrict__ p, int n) {
    int i = blockIdx.x * 256 + threadIdx.x;
    if (i < n) p[i] = 0.f;
}

// ---------------------------------------------------------------------------
// Kernel 5b: A[b,d] += sum_s Khat*Vhat over an s-chunk (atomic f32)
// grid (DIM/256, SEQ/256, BATCH)
// ---------------------------------------------------------------------------
__global__ void __launch_bounds__(256)
k_accum_A(const float* __restrict__ Kf, const float* __restrict__ Vf,
          const float* __restrict__ rnorm, float* __restrict__ Aacc) {
    const int b  = blockIdx.z;
    const int d  = blockIdx.x * 256 + threadIdx.x;
    const int s0 = blockIdx.y * 256;
    float acc = 0.f;
    for (int s = s0; s < s0 + 256; ++s) {
        size_t r = (size_t)b * SEQ + s;
        float rk = rnorm[r];
        float rv = rnorm[MROWS + r];
        acc += (Kf[r * DIM + d] * rk) * (Vf[r * DIM + d] * rv);
    }
    atomicAdd(&Aacc[b * DIM + d], acc);
}

// ---------------------------------------------------------------------------
// Kernel 6: Q *= (0.5*A + 0.5), bf16 in place
// ---------------------------------------------------------------------------
__global__ void __launch_bounds__(256)
k_scale_q(unsigned short* __restrict__ Qbf, const float* __restrict__ Aacc) {
    size_t i = (size_t)blockIdx.x * 256 + threadIdx.x;   // over MROWS*DIM
    int d   = (int)(i & (DIM - 1));
    int row = (int)(i >> 11);                            // /DIM
    int b   = row >> 12;                                 // /SEQ
    float a = 0.5f * Aacc[b * DIM + d] + 0.5f;
    Qbf[i] = f2bf(bf2f(Qbf[i]) * a);
}

// ---------------------------------------------------------------------------
// Host-side orchestration
// ---------------------------------------------------------------------------
extern "C" void kernel_launch(void* const* d_in, const int* in_sizes, int n_in,
                              void* d_out, int out_size, void* d_ws, size_t ws_size,
                              hipStream_t stream) {
    (void)in_sizes; (void)n_in; (void)out_size; (void)ws_size;

    const float* X     = (const float*)d_in[0];
    const float* W_in  = (const float*)d_in[1];
    const float* b_in  = (const float*)d_in[2];
    const float* W_out = (const float*)d_in[3];
    const float* b_out = (const float*)d_in[4];
    const float* gamma = (const float*)d_in[5];
    const float* beta  = (const float*)d_in[6];
    float* Y = (float*)d_out;

    // workspace layout
    char* ws = (char*)d_ws;
    size_t off = 0;
    unsigned short* Xn    = (unsigned short*)(ws + off); off += (size_t)MROWS * DIM * 2;
    unsigned short* WinT  = (unsigned short*)(ws + off); off += (size_t)N3 * DIM * 2;
    unsigned short* WoutT = (unsigned short*)(ws + off); off += (size_t)DIM * DIM * 2;
    unsigned short* Qbf   = (unsigned short*)(ws + off); off += (size_t)MROWS * DIM * 2;
    float* Kf    = (float*)(ws + off); off += (size_t)MROWS * DIM * 4;
    float* Vf    = (float*)(ws + off); off += (size_t)MROWS * DIM * 4;
    float* rnorm = (float*)(ws + off); off += (size_t)2 * MROWS * 4;
    float* Aacc  = (float*)(ws + off); off += (size_t)BATCH * DIM * 4;

    // 1. weights -> bf16, pre-transposed to N x K
    {
        size_t n = (size_t)DIM * N3;
        k_wtrans_bf16<<<dim3((unsigned)((n + 255) / 256)), 256, 0, stream>>>(W_in, WinT, DIM, N3);
        size_t m = (size_t)DIM * DIM;
        k_wtrans_bf16<<<dim3((unsigned)((m + 255) / 256)), 256, 0, stream>>>(W_out, WoutT, DIM, DIM);
    }
    // 2. layernorm -> Xn bf16
    k_layernorm<<<dim3(MROWS), 256, 0, stream>>>(X, gamma, beta, Xn);
    // 3. QKV GEMM (WMMA bf16 + async double-buffer), epilogue: +b_in, split Q/K/V
    k_gemm_wmma<<<dim3(N3 / BN, MROWS / BM), 256, 0, stream>>>(
        Xn, WinT, b_in, MROWS, N3, DIM, /*mode=*/0,
        Qbf, Kf, Vf, nullptr, nullptr);
    // 4. rotary + row norms on K and V
    k_rotary_norm<<<dim3(MROWS, 2), 256, 0, stream>>>(Kf, Vf, rnorm);
    // 5. A = sum_s cos_sim(K)*cos_sim(V)
    k_zero<<<dim3((BATCH * DIM + 255) / 256), 256, 0, stream>>>(Aacc, BATCH * DIM);
    k_accum_A<<<dim3(DIM / 256, SEQ / 256, BATCH), 256, 0, stream>>>(Kf, Vf, rnorm, Aacc);
    // 6. Q *= 0.5*A + 0.5
    k_scale_q<<<dim3((unsigned)(((size_t)MROWS * DIM) / 256)), 256, 0, stream>>>(Qbf, Aacc);
    // 7. output GEMM (WMMA bf16 + async double-buffer), epilogue: +b_out + X -> Y
    k_gemm_wmma<<<dim3(DIM / BN, MROWS / BM), 256, 0, stream>>>(
        Qbf, WoutT, b_out, MROWS, DIM, DIM, /*mode=*/1,
        nullptr, nullptr, nullptr, X, Y);
}
